// SelfAttention_26259430048753
// MI455X (gfx1250) — matrix operand order repaired and checked by reading, not compiled, not run on hardware
//
#include <hip/hip_runtime.h>
#include <math.h>

typedef __attribute__((ext_vector_type(16))) _Float16 v16h;
typedef __attribute__((ext_vector_type(8)))  _Float16 v8h;
typedef __attribute__((ext_vector_type(8)))  float    v8f;
typedef __attribute__((ext_vector_type(4)))  float    v4f;
typedef __attribute__((ext_vector_type(4)))  int      v4i;

#define B_   2
#define S_   2048
#define DIM_ 4096
#define NH_  32
#define NKV_ 8
#define HD_  128

// ---- CDNA5 async global->LDS path (guarded; falls back to sync copy) ------
#if defined(__has_builtin)
#if __has_builtin(__builtin_amdgcn_global_load_async_to_lds_b128) && \
    __has_builtin(__builtin_amdgcn_s_wait_asynccnt)
#define HAVE_ASYNC_LDS 1
#endif
#endif

#define AS1_ __attribute__((address_space(1)))
#define AS3_ __attribute__((address_space(3)))

__device__ __forceinline__ void cp16_g2l(_Float16* dst_lds, const _Float16* src_g) {
#if defined(HAVE_ASYNC_LDS)
  __builtin_amdgcn_global_load_async_to_lds_b128(
      (AS1_ v4i*)(void*)const_cast<_Float16*>(src_g),
      (AS3_ v4i*)(void*)dst_lds, 0, 0);
#else
  *(v8h*)dst_lds = *(const v8h*)src_g;
#endif
}

__device__ __forceinline__ void stage_fence() {
#if defined(HAVE_ASYNC_LDS)
  __builtin_amdgcn_s_wait_asynccnt(0);
#endif
  __syncthreads();
}

// ---------------------------------------------------------------------------
// Elementwise fp32 -> f16 convert, 8 elements / thread
// ---------------------------------------------------------------------------
__global__ void cvt_f32_f16_x8(const float* __restrict__ in,
                               _Float16* __restrict__ out, size_t n8) {
  size_t i = (size_t)blockIdx.x * blockDim.x + threadIdx.x;
  if (i >= n8) return;
  v8f x = ((const v8f*)in)[i];
  v8h y;
#pragma unroll
  for (int j = 0; j < 8; ++j) y[j] = (_Float16)x[j];
  ((v8h*)out)[i] = y;
}

// ---------------------------------------------------------------------------
// RoPE on fp32 (B,S,H,HD) -> f16 same layout; 4 pairs (8 elems) / thread
// ---------------------------------------------------------------------------
__global__ void rope_cvt_x8(const float* __restrict__ in,
                            const float* __restrict__ cosT,
                            const float* __restrict__ sinT,
                            _Float16* __restrict__ out, int H, size_t n) {
  size_t idx = (size_t)blockIdx.x * blockDim.x + threadIdx.x;
  if (idx >= n) return;
  int seg = (int)(idx & 15);        // 16 segments of 8 elements over HD=128
  size_t t = idx >> 4;
  int h = (int)(t % H);
  size_t bs = t / H;                // b*S + s
  int s = (int)(bs & (S_ - 1));
  size_t base = (bs * (size_t)H + h) * HD_ + seg * 8;
  v8f xv = *(const v8f*)(in + base);
  v4f c  = *(const v4f*)(cosT + (size_t)s * 64 + seg * 4);
  v4f sn = *(const v4f*)(sinT + (size_t)s * 64 + seg * 4);
  v8h o;
#pragma unroll
  for (int j = 0; j < 4; ++j) {
    float xr = xv[2 * j], xi = xv[2 * j + 1];
    o[2 * j]     = (_Float16)(xr * c[j] - xi * sn[j]);
    o[2 * j + 1] = (_Float16)(xr * sn[j] + xi * c[j]);
  }
  *(v8h*)(out + base) = o;
}

// ---------------------------------------------------------------------------
// WMMA GEMM (compile-time N,K): C[M][N] = A[M][K] * B[K][N]
// A,B f16 row-major; C f32 row-major. 256 threads = 8 waves;
// block tile 128x128; wave tile 64x32 (4x2 WMMA accumulators); K-step 32.
// ---------------------------------------------------------------------------
template <int N, int K>
__global__ __launch_bounds__(256) void gemm_f16_wmma(
    const _Float16* __restrict__ A, const _Float16* __restrict__ B,
    float* __restrict__ C) {
  __shared__ __align__(16) _Float16 sA[128 * 40];   // [row][k] padded
  __shared__ __align__(16) _Float16 sBt[128 * 40];  // [n][k]  (transposed)

  const int tid  = threadIdx.x;
  const int lane = tid & 31;
  const int wid  = tid >> 5;
  const int lm   = lane & 15;
  const int half = lane >> 4;
  const int m0 = blockIdx.y * 128;
  const int n0 = blockIdx.x * 128;
  const int wrow = wid >> 2;   // 0..1
  const int wcol = wid & 3;    // 0..3

  const v8f vzero = {0.f, 0.f, 0.f, 0.f, 0.f, 0.f, 0.f, 0.f};
  v8f acc[4][2];
#pragma unroll
  for (int mt = 0; mt < 4; ++mt)
#pragma unroll
    for (int nt = 0; nt < 2; ++nt) acc[mt][nt] = vzero;

  // per-thread staging coordinates
  const int arow = tid >> 1, aseg = (tid & 1) * 16;       // A: 128x32
  const int bkk = tid >> 3, bnseg = (tid & 7) * 16;       // B: 32x128

  for (int k0 = 0; k0 < K; k0 += 32) {
    // --- stage A tile 128x32 (contiguous; async global->LDS) ---
    {
      const _Float16* g = A + (size_t)(m0 + arow) * K + k0 + aseg;
      cp16_g2l(&sA[arow * 40 + aseg], g);
      cp16_g2l(&sA[arow * 40 + aseg + 8], g + 8);
    }
    // --- stage B tile 32x128, transposed into sBt[n][k] ---
    {
      const _Float16* g = B + (size_t)(k0 + bkk) * N + n0 + bnseg;
      v8h lo = *(const v8h*)g;
      v8h hi = *(const v8h*)(g + 8);
#pragma unroll
      for (int j = 0; j < 8; ++j) sBt[(bnseg + j) * 40 + bkk]     = lo[j];
#pragma unroll
      for (int j = 0; j < 8; ++j) sBt[(bnseg + 8 + j) * 40 + bkk] = hi[j];
    }
    if (k0 + 32 < K) {  // prefetch next K-step tiles
      __builtin_prefetch(A + (size_t)(m0 + arow) * K + k0 + 32 + aseg, 0, 3);
      __builtin_prefetch(B + (size_t)(k0 + 32 + bkk) * N + n0 + bnseg, 0, 3);
    }
    stage_fence();

    // --- fragments ---
    v16h af[4];
#pragma unroll
    for (int mt = 0; mt < 4; ++mt) {
      const _Float16* p = &sA[(wrow * 64 + mt * 16 + lm) * 40 + half * 8];
      v8h lo = *(const v8h*)p;
      v8h hi = *(const v8h*)(p + 16);
#pragma unroll
      for (int i = 0; i < 8; ++i) { af[mt][i] = lo[i]; af[mt][8 + i] = hi[i]; }
    }
    v16h bf[2];
#pragma unroll
    for (int nt = 0; nt < 2; ++nt) {
      const _Float16* p = &sBt[(wcol * 32 + nt * 16 + lm) * 40 + half * 8];
      v8h lo = *(const v8h*)p;
      v8h hi = *(const v8h*)(p + 16);
#pragma unroll
      for (int i = 0; i < 8; ++i) { bf[nt][i] = lo[i]; bf[nt][8 + i] = hi[i]; }
    }
#pragma unroll
    for (int mt = 0; mt < 4; ++mt)
#pragma unroll
      for (int nt = 0; nt < 2; ++nt)
        acc[mt][nt] = __builtin_amdgcn_wmma_f32_16x16x32_f16(
            false, af[mt], false, bf[nt], (short)0, acc[mt][nt], false, false);
    __syncthreads();
  }

  // --- epilogue: constant strides -> immediate-offset stores ---
#pragma unroll
  for (int mt = 0; mt < 4; ++mt) {
    float* cp = C + (size_t)(m0 + wrow * 64 + mt * 16 + half * 8) * N +
                n0 + wcol * 32 + lm;
#pragma unroll
    for (int nt = 0; nt < 2; ++nt)
#pragma unroll
      for (int r = 0; r < 8; ++r) cp[r * N + nt * 16] = acc[mt][nt][r];
  }
}

// ---------------------------------------------------------------------------
// Flash attention (no mask): one block = (b, head, 128 q rows).
// Each wave owns 16 q rows; 8 waves sweep K together sharing K/V tiles.
// ---------------------------------------------------------------------------
__global__ __launch_bounds__(256) void attn_wmma(
    const _Float16* __restrict__ Q, const _Float16* __restrict__ K,
    const _Float16* __restrict__ V, float* __restrict__ O) {
  __shared__ __align__(16) _Float16 sK[32 * 136];    // [kk][d] row-major
  __shared__ __align__(16) _Float16 sVt[128 * 40];   // [d][kk] transposed
  __shared__ __align__(16) _Float16 sP[8 * 16 * 40]; // per-wave P staging

  const int tid  = threadIdx.x;
  const int lane = tid & 31;
  const int wid  = tid >> 5;
  const int lm   = lane & 15;
  const int half = lane >> 4;

  int bid = blockIdx.x;
  int qt  = bid & 15;          // S/128 q-tiles
  int h   = (bid >> 4) & 31;   // head
  int b   = bid >> 9;
  int kvh = h >> 2;            // n_rep = NH/NKV = 4

  const int s0 = qt * 128 + wid * 16;

  // Q fragments in registers: 4 chunks of contraction-dim 32 over HD=128
  v16h qf[4];
  {
    const size_t qbase = (((size_t)b * S_ + s0 + lm) * NH_ + h) * HD_;
#pragma unroll
    for (int c = 0; c < 4; ++c) {
      const _Float16* p = Q + qbase + c * 32 + half * 8;
      v8h lo = *(const v8h*)p;
      v8h hi = *(const v8h*)(p + 16);
#pragma unroll
      for (int i = 0; i < 8; ++i) { qf[c][i] = lo[i]; qf[c][8 + i] = hi[i]; }
    }
  }

  const v8f vzero = {0.f, 0.f, 0.f, 0.f, 0.f, 0.f, 0.f, 0.f};
  v8f acc[8];
#pragma unroll
  for (int t = 0; t < 8; ++t) acc[t] = vzero;
  float mrow[8], lrow[8];
#pragma unroll
  for (int r = 0; r < 8; ++r) { mrow[r] = -1e30f; lrow[r] = 0.f; }

  const float scale = 0.08838834764831845f;  // 1/sqrt(128)
  _Float16* myP = &sP[wid * 16 * 40];

  const int kk   = tid >> 3;        // 0..31
  const int dseg = (tid & 7) * 16;  // 0..112

  for (int k0 = 0; k0 < S_; k0 += 32) {
    // --- stage K tile (async, row-major) and V tile (transposed) ---
    {
      size_t gb = (((size_t)b * S_ + k0 + kk) * NKV_ + kvh) * HD_ + dseg;
      cp16_g2l(&sK[kk * 136 + dseg], K + gb);
      cp16_g2l(&sK[kk * 136 + dseg + 8], K + gb + 8);
      v8h vlo = *(const v8h*)(V + gb);
      v8h vhi = *(const v8h*)(V + gb + 8);
#pragma unroll
      for (int j = 0; j < 8; ++j) sVt[(dseg + j) * 40 + kk]     = vlo[j];
#pragma unroll
      for (int j = 0; j < 8; ++j) sVt[(dseg + 8 + j) * 40 + kk] = vhi[j];
    }
    stage_fence();

    // --- scores: two 16x16 tiles (k cols +0..15, +16..31) ---
    v8f sc[2];
#pragma unroll
    for (int j = 0; j < 2; ++j) {
      sc[j] = vzero;
      int kc = j * 16 + lm;
#pragma unroll
      for (int c = 0; c < 4; ++c) {
        const _Float16* p = &sK[kc * 136 + c * 32 + half * 8];
        v8h lo = *(const v8h*)p;
        v8h hi = *(const v8h*)(p + 16);
        v16h bfr;
#pragma unroll
        for (int i = 0; i < 8; ++i) { bfr[i] = lo[i]; bfr[8 + i] = hi[i]; }
        sc[j] = __builtin_amdgcn_wmma_f32_16x16x32_f16(
            false, qf[c], false, bfr, (short)0, sc[j], false, false);
      }
    }

    // --- online softmax: row max within 16-lane column group ---
    float alpha[8];
#pragma unroll
    for (int r = 0; r < 8; ++r) {
      float v = fmaxf(sc[0][r], sc[1][r]) * scale;
      v = fmaxf(v, __shfl_xor(v, 1, 32));
      v = fmaxf(v, __shfl_xor(v, 2, 32));
      v = fmaxf(v, __shfl_xor(v, 4, 32));
      v = fmaxf(v, __shfl_xor(v, 8, 32));
      float mn = fmaxf(mrow[r], v);
      alpha[r] = __expf(mrow[r] - mn);
      mrow[r]  = mn;
    }

    // --- P = exp(scale*s - m), stage to wave-private LDS, row sums ---
    float rsum[8];
#pragma unroll
    for (int r = 0; r < 8; ++r) rsum[r] = 0.f;
#pragma unroll
    for (int j = 0; j < 2; ++j) {
#pragma unroll
      for (int r = 0; r < 8; ++r) {
        float p = __expf(sc[j][r] * scale - mrow[r]);
        rsum[r] += p;
        myP[(half * 8 + r) * 40 + j * 16 + lm] = (_Float16)p;
      }
    }
#pragma unroll
    for (int r = 0; r < 8; ++r) {
      float v = rsum[r];
      v += __shfl_xor(v, 1, 32);
      v += __shfl_xor(v, 2, 32);
      v += __shfl_xor(v, 4, 32);
      v += __shfl_xor(v, 8, 32);
      lrow[r] = lrow[r] * alpha[r] + v;
    }
#pragma unroll
    for (int t = 0; t < 8; ++t)
#pragma unroll
      for (int r = 0; r < 8; ++r) acc[t][r] *= alpha[r];

    // --- reload P as A-fragment (16x32) ---
    asm volatile("s_wait_dscnt 0" ::: "memory");
    v16h pf;
    {
      const _Float16* p = &myP[lm * 40 + half * 8];
      v8h lo = *(const v8h*)p;
      v8h hi = *(const v8h*)(p + 16);
#pragma unroll
      for (int i = 0; i < 8; ++i) { pf[i] = lo[i]; pf[8 + i] = hi[i]; }
    }

    // --- O += P * V : 8 output N-tiles of 16 ---
#pragma unroll
    for (int nt = 0; nt < 8; ++nt) {
      const _Float16* p = &sVt[(nt * 16 + lm) * 40 + half * 8];
      v8h lo = *(const v8h*)p;
      v8h hi = *(const v8h*)(p + 16);
      v16h bfr;
#pragma unroll
      for (int i = 0; i < 8; ++i) { bfr[i] = lo[i]; bfr[8 + i] = hi[i]; }
      acc[nt] = __builtin_amdgcn_wmma_f32_16x16x32_f16(
          false, pf, false, bfr, (short)0, acc[nt], false, false);
    }
    __syncthreads();
  }

  // --- normalize and store (b, s, h, d) fp32; constant strides ---
  float* op = O + (((size_t)b * S_ + s0 + half * 8) * NH_ + h) * HD_ + lm;
#pragma unroll
  for (int nt = 0; nt < 8; ++nt)
#pragma unroll
    for (int r = 0; r < 8; ++r)
      op[(size_t)r * (NH_ * HD_) + nt * 16] = acc[nt][r] / lrow[r];
}

// ---------------------------------------------------------------------------
// Host-side launch
// ---------------------------------------------------------------------------
extern "C" void kernel_launch(void* const* d_in, const int* in_sizes, int n_in,
                              void* d_out, int out_size, void* d_ws,
                              size_t ws_size, hipStream_t stream) {
  const float* x    = (const float*)d_in[0];
  // d_in[1] = start_pos (0), unused
  const float* fcos = (const float*)d_in[2];
  const float* fsin = (const float*)d_in[3];
  const float* wq   = (const float*)d_in[4];
  const float* wk   = (const float*)d_in[5];
  const float* wv   = (const float*)d_in[6];
  const float* wo   = (const float*)d_in[7];
  float* out = (float*)d_out;

  char* ws = (char*)d_ws;
  const size_t NX = (size_t)B_ * S_ * DIM_;           // 16,777,216
  const size_t NK = (size_t)B_ * S_ * NKV_ * HD_;     //  4,194,304

  _Float16* Xh = (_Float16*)(ws);                     // 33.5 MB (later: Oh)
  _Float16* Wh = (_Float16*)(ws + 33554432ull);       // 33.5 MB (reused per weight)
  float*    Qf = (float*)(ws + 67108864ull);          // 67 MB (later: attn out f32)
  float*    Kf = (float*)(ws + 134217728ull);         // 16.8 MB
  float*    Vf = (float*)(ws + 150994944ull);         // 16.8 MB
  _Float16* Qh = (_Float16*)(ws + 167772160ull);      // 33.5 MB
  _Float16* Kh = (_Float16*)(ws + 201326592ull);      // 8.4 MB
  _Float16* Vh = (_Float16*)(ws + 209715200ull);      // 8.4 MB

  auto blks = [](size_t n) { return (unsigned)((n + 255) / 256); };

  // x -> f16
  cvt_f32_f16_x8<<<blks(NX / 8), 256, 0, stream>>>(x, Xh, NX / 8);

  // Q = X @ Wq   (4096 x 4096 x 4096)
  cvt_f32_f16_x8<<<blks(NX / 8), 256, 0, stream>>>(wq, Wh, NX / 8);
  gemm_f16_wmma<4096, 4096><<<dim3(32, 32), 256, 0, stream>>>(Xh, Wh, Qf);
  // K = X @ Wk   (4096 x 1024 x 4096)
  cvt_f32_f16_x8<<<blks(NK / 8), 256, 0, stream>>>(wk, Wh, NK / 8);
  gemm_f16_wmma<1024, 4096><<<dim3(8, 32), 256, 0, stream>>>(Xh, Wh, Kf);
  // V = X @ Wv
  cvt_f32_f16_x8<<<blks(NK / 8), 256, 0, stream>>>(wv, Wh, NK / 8);
  gemm_f16_wmma<1024, 4096><<<dim3(8, 32), 256, 0, stream>>>(Xh, Wh, Vf);

  // RoPE(Q) -> f16, RoPE(K) -> f16, V -> f16
  const size_t nQ8 = (size_t)B_ * S_ * NH_ * 16;   // 8-elem chunks
  const size_t nK8 = (size_t)B_ * S_ * NKV_ * 16;
  rope_cvt_x8<<<blks(nQ8), 256, 0, stream>>>(Qf, fcos, fsin, Qh, NH_, nQ8);
  rope_cvt_x8<<<blks(nK8), 256, 0, stream>>>(Kf, fcos, fsin, Kh, NKV_, nK8);
  cvt_f32_f16_x8<<<blks(NK / 8), 256, 0, stream>>>(Vf, Vh, NK / 8);

  // attention -> fp32 (reuse Qf region)
  attn_wmma<<<B_ * NH_ * (S_ / 128), 256, 0, stream>>>(Qh, Kh, Vh, Qf);

  // attn out -> f16 (reuse Xh region), final projection into d_out
  cvt_f32_f16_x8<<<blks(NX / 8), 256, 0, stream>>>(Qf, Xh, NX / 8);
  cvt_f32_f16_x8<<<blks(NX / 8), 256, 0, stream>>>(wo, Wh, NX / 8);
  gemm_f16_wmma<4096, 4096><<<dim3(32, 32), 256, 0, stream>>>(Xh, Wh, out);
}
